// TrilinearAttention_12799002542454
// MI455X (gfx1250) — matrix operand order repaired and checked by reading, not compiled, not run on hardware
//
#include <hip/hip_runtime.h>
#include <hip/hip_bf16.h>

// ---------------------------------------------------------------------------
// TrilinearAttention on MI455X (gfx1250), wave32 + v_wmma_f32_16x16x32_bf16.
// B=8, N=8, P=400, Q=64, D=1024.  Softmax over Q cancels A/bias terms.
// ---------------------------------------------------------------------------

typedef __attribute__((ext_vector_type(16))) __bf16 v16bf;
typedef __attribute__((ext_vector_type(8)))  __bf16 v8bf;
typedef __attribute__((ext_vector_type(4)))  __bf16 v4bf;
typedef __attribute__((ext_vector_type(8)))  float  v8f;

#define B_   8
#define N_   8
#define P_   400
#define Q_   64
#define D_   1024
#define NEG_INF_ -1e20f

#define CAT16(lo, hi) __builtin_shufflevector((lo), (hi), 0,1,2,3,4,5,6,7,8,9,10,11,12,13,14,15)

// ---------------------------------------------------------------------------
// Prep: per (b,q): qlog = mask ? questions[b,q,:]·Wq + bq : -1e20
//       qbf[b][q][d]  = bf16(questions)            (row-major, GEMM1 B-frags)
//       qT [b][d][q]  = bf16(questions) transposed (GEMM2 B-frags)
// ---------------------------------------------------------------------------
__global__ void __launch_bounds__(256)
prep_q_kernel(const float* __restrict__ questions,
              const float* __restrict__ Wq,
              const float* __restrict__ bq,
              const int*   __restrict__ qmask,
              float*  __restrict__ qlog,
              __bf16* __restrict__ qbf,
              __bf16* __restrict__ qT) {
    const int bqi = blockIdx.x;          // 0..511
    const int b = bqi >> 6;
    const int q = bqi & 63;
    const int t = threadIdx.x;           // 0..255, 4 elems each

    const float* row = questions + (b * Q_ + q) * D_;
    float4 x  = ((const float4*)row)[t];
    float4 wv = ((const float4*)Wq)[t];

    v4bf xb;
    xb[0] = (__bf16)x.x; xb[1] = (__bf16)x.y; xb[2] = (__bf16)x.z; xb[3] = (__bf16)x.w;
    *((v4bf*)(qbf + (size_t)(b * Q_ + q) * D_ + t * 4)) = xb;   // coalesced 8B

    __bf16* qTc = qT + (size_t)b * D_ * Q_ + q;                 // column q
    qTc[(t * 4 + 0) * Q_] = xb[0];
    qTc[(t * 4 + 1) * Q_] = xb[1];
    qTc[(t * 4 + 2) * Q_] = xb[2];
    qTc[(t * 4 + 3) * Q_] = xb[3];

    float part = x.x * wv.x + x.y * wv.y + x.z * wv.z + x.w * wv.w;
    __shared__ float red[256];
    red[t] = part;
    __syncthreads();
    for (int s = 128; s > 0; s >>= 1) {
        if (t < s) red[t] += red[t + s];
        __syncthreads();
    }
    if (t == 0)
        qlog[b * Q_ + q] = (qmask[b * Q_ + q] != 0) ? (red[0] + bq[0]) : NEG_INF_;
}

// ---------------------------------------------------------------------------
// Main: one wave (32 lanes) per 16-row paragraph tile.
//   GEMM1: logits[16x64] = (para_tile * w) @ questions^T   (bf16 WMMA, f32 acc)
//   softmax over Q (+ qlog, mask already folded into qlog)
//   GEMM2: c2q[16x1024] = att @ questions                  (bf16 WMMA, f32 acc)
//   fused output: [para | c2q | c2q*para] with 16B/lane, 256B/half-wave stores
// ---------------------------------------------------------------------------
__global__ void __launch_bounds__(32)
trilinear_attn_kernel(const float*  __restrict__ paragraphs,
                      const float*  __restrict__ w,
                      const float*  __restrict__ qlog,
                      const __bf16* __restrict__ qbf,
                      const __bf16* __restrict__ qT,
                      float* __restrict__ out) {
    const int tile = blockIdx.x;               // 0..1599
    const int b  = tile / (N_ * 25);
    const int n  = (tile / 25) % N_;
    const int p0 = (tile % 25) * 16;

    const int lane = threadIdx.x;              // wave32
    const int half = lane >> 4;                // 0/1 (K-interleave half)
    const int mr   = lane & 15;                // row (A/M) or col (B/N) index
    const int koff = 8 * half;

    __shared__ __align__(16) float  wlds[D_];
    __shared__ __align__(16) float  fbuf[16 * Q_];   // logits, then c2q chunks
    __shared__ __align__(16) __bf16 attb[16 * Q_];

    // stage w into LDS (32 lanes x 8 float4 = 1024 floats)
#pragma unroll
    for (int i = 0; i < 8; ++i)
        ((float4*)wlds)[lane * 8 + i] = ((const float4*)w)[lane * 8 + i];
    __syncthreads();

    // ---------------- GEMM1: 16x64 logits over K = D ----------------
    const float* prow = paragraphs + (size_t)((b * N_ + n) * P_ + p0 + mr) * D_;

    v8f zero = 0.0f;
    v8f acc[4];
#pragma unroll
    for (int qt = 0; qt < 4; ++qt) acc[qt] = zero;

    for (int kb = 0; kb < D_; kb += 32) {
        // A-frag: lanes 0-15 get K {kb+0..7, kb+16..23}; lanes 16-31 +8
        const float* pA = prow + kb + koff;
        float4 a0 = ((const float4*)pA)[0];
        float4 a1 = ((const float4*)pA)[1];
        float4 a2 = ((const float4*)(pA + 16))[0];
        float4 a3 = ((const float4*)(pA + 16))[1];
        __builtin_prefetch(pA + 128, 0, 0);     // global_prefetch_b8

        const float* wp = wlds + kb + koff;
        float4 w0 = ((const float4*)wp)[0];
        float4 w1 = ((const float4*)wp)[1];
        float4 w2 = ((const float4*)(wp + 16))[0];
        float4 w3 = ((const float4*)(wp + 16))[1];

        float av[16] = {a0.x*w0.x, a0.y*w0.y, a0.z*w0.z, a0.w*w0.w,
                        a1.x*w1.x, a1.y*w1.y, a1.z*w1.z, a1.w*w1.w,
                        a2.x*w2.x, a2.y*w2.y, a2.z*w2.z, a2.w*w2.w,
                        a3.x*w3.x, a3.y*w3.y, a3.z*w3.z, a3.w*w3.w};
        v16bf A;
#pragma unroll
        for (int i = 0; i < 16; ++i) A[i] = (__bf16)av[i];

        // B-frags: 4 Q-tiles; per lane K {kb+koff+0..7, kb+koff+16..23} of qbf row
#pragma unroll
        for (int qt = 0; qt < 4; ++qt) {
            const __bf16* qp = qbf + (size_t)(b * Q_ + qt * 16 + mr) * D_ + kb + 8 * half;
            v8bf b0 = *(const v8bf*)qp;
            v8bf b1 = *(const v8bf*)(qp + 16);
            v16bf Bf = CAT16(b0, b1);
            acc[qt] = __builtin_amdgcn_wmma_f32_16x16x32_bf16(
                false, A, false, Bf, (short)0, acc[qt], false, false);
        }
    }

    // spill logits to LDS: VGPR v of lane -> (m = v + 8*half, q = qt*16 + mr)
#pragma unroll
    for (int qt = 0; qt < 4; ++qt)
#pragma unroll
        for (int v = 0; v < 8; ++v)
            fbuf[(v + 8 * half) * Q_ + qt * 16 + mr] = acc[qt][v];
    __syncthreads();

    // ---------------- softmax over Q (lanes 0-15, one row each) ------------
    if (lane < 16) {
        float* lrow = fbuf + lane * Q_;
        const float* ql = qlog + b * Q_;
        float mx = -3.4e38f;
        for (int q = 0; q < Q_; ++q) {
            float l = lrow[q] + ql[q];          // masked positions: ql = -1e20
            lrow[q] = l;
            mx = fmaxf(mx, l);
        }
        float sum = 0.f;
        for (int q = 0; q < Q_; ++q) {
            float e = __expf(lrow[q] - mx);
            lrow[q] = e;
            sum += e;
        }
        float inv = 1.f / sum;
        for (int q = 0; q < Q_; ++q)
            attb[lane * Q_ + q] = (__bf16)(lrow[q] * inv);
    }
    __syncthreads();

    // ---------------- GEMM2: c2q = att[16x64] @ questions[64xD] ------------
    // A2-frags from LDS att (two K=32 frags)
    v16bf A2[2];
#pragma unroll
    for (int f = 0; f < 2; ++f) {
        const __bf16* ap = attb + mr * Q_ + f * 32 + koff;
        v8bf r0 = *(const v8bf*)ap;             // K = f*32 + koff + 0..7
        v8bf r1 = *(const v8bf*)(ap + 16);      // K = f*32 + koff + 16..23
        A2[f] = CAT16(r0, r1);
    }

    const float* ptile = paragraphs + (size_t)((b * N_ + n) * P_ + p0) * D_;
    float*       otile = out        + (size_t)((b * N_ + n) * P_ + p0) * (3 * D_);
    const int cb = mr * 4;                      // column-of-chunk for epilogue

    for (int dc = 0; dc < D_; dc += 64) {       // 64 output cols / chunk
        v8f c[4];
#pragma unroll
        for (int nt = 0; nt < 4; ++nt) c[nt] = zero;

#pragma unroll
        for (int nt = 0; nt < 4; ++nt) {
            const int d0 = dc + nt * 16 + mr;
            const __bf16* qp = qT + ((size_t)b * D_ + d0) * Q_ + 8 * half;
#pragma unroll
            for (int f = 0; f < 2; ++f) {
                v8bf b0 = *(const v8bf*)(qp + f * 32);
                v8bf b1 = *(const v8bf*)(qp + f * 32 + 16);
                v16bf Bf = CAT16(b0, b1);
                c[nt] = __builtin_amdgcn_wmma_f32_16x16x32_bf16(
                    false, A2[f], false, Bf, (short)0, c[nt], false, false);
            }
        }

        // spill c2q chunk to LDS (accumulator layout -> row-major 16x64)
#pragma unroll
        for (int nt = 0; nt < 4; ++nt)
#pragma unroll
            for (int v = 0; v < 8; ++v)
                fbuf[(v + 8 * half) * 64 + nt * 16 + mr] = c[nt][v];
        __syncthreads();

        // fused concat write, fully vectorized: 16 consecutive lanes write a
        // 256B-contiguous row segment per region (global_store_b128).
#pragma unroll
        for (int r = 0; r < 8; ++r) {
            const int row = 8 * half + r;
            float4 cc = *(const float4*)&fbuf[row * 64 + cb];
            float4 pv = *(const float4*)&ptile[(size_t)row * D_ + dc + cb];
            float* orow = otile + (size_t)row * (3 * D_) + dc + cb;
            float4 cp = make_float4(cc.x * pv.x, cc.y * pv.y,
                                    cc.z * pv.z, cc.w * pv.w);
            *(float4*)(orow)            = pv;   // paragraphs
            *(float4*)(orow + D_)       = cc;   // c2q
            *(float4*)(orow + 2 * D_)   = cp;   // c2q * paragraphs
        }
        __syncthreads();                        // LDS reused next chunk
    }
}

// ---------------------------------------------------------------------------
extern "C" void kernel_launch(void* const* d_in, const int* in_sizes, int n_in,
                              void* d_out, int out_size, void* d_ws, size_t ws_size,
                              hipStream_t stream) {
    const float* questions  = (const float*)d_in[0];   // [8,64,1024]
    const float* paragraphs = (const float*)d_in[1];   // [8,8,400,1024]
    const float* w          = (const float*)d_in[2];   // [1024]
    // d_in[3] bias, d_in[4] Wp, d_in[5] bp: dead (softmax shift-invariance)
    const float* Wq         = (const float*)d_in[6];   // [1024]
    const float* bq         = (const float*)d_in[7];   // [1]
    const int*   qmask      = (const int*)d_in[8];     // [8,64]
    float* out = (float*)d_out;

    char* ws = (char*)d_ws;
    float*  qlog = (float*)ws;                                   // 2 KB
    __bf16* qbf  = (__bf16*)(ws + 4096);                         // 1 MB
    __bf16* qT   = (__bf16*)(ws + 4096 + (1u << 20));            // 1 MB

    prep_q_kernel<<<B_ * Q_, 256, 0, stream>>>(questions, Wq, bq, qmask,
                                               qlog, qbf, qT);
    trilinear_attn_kernel<<<B_ * N_ * (P_ / 16), 32, 0, stream>>>(
        paragraphs, w, qlog, qbf, qT, out);
}
